// MPSRecurrence_49392123904227
// MI455X (gfx1250) — compile-verified, matrix operand order checked by reading, not run on hardware
//
#include <hip/hip_runtime.h>
#include <cstdint>

// ---------------------------------------------------------------- constants
static constexpr int Bc = 4;
static constexpr int Tc = 8192;
static constexpr int Dc = 1024;   // D_IN
static constexpr int Hc = 1024;   // D_H
static constexpr int Oc = 1024;   // D_OUT
static constexpr int Mc = Bc * Tc;          // 32768 rows
static constexpr int NC = 64;               // scan chunks
static constexpr int CL = Tc / NC;          // 128 steps per chunk
static constexpr float LN_EPS = 1e-5f;
static constexpr int LROW = Dc + 8;         // padded LDS row (elements) to avoid bank conflicts

// ws layout (bytes)
static constexpr size_t OFF_XB     = 0;
static constexpr size_t SZ_XB      = (size_t)Mc * Dc * 2;
static constexpr size_t OFF_WIHB   = OFF_XB + SZ_XB;
static constexpr size_t SZ_W       = (size_t)Hc * Dc * 2;
static constexpr size_t OFF_WGB    = OFF_WIHB + SZ_W;
static constexpr size_t OFF_WOB    = OFF_WGB + SZ_W;
static constexpr size_t OFF_ALPHA  = OFF_WOB + SZ_W;
static constexpr size_t SZ_BTH     = (size_t)Mc * Hc * 4;
static constexpr size_t OFF_BETA   = OFF_ALPHA + SZ_BTH;   // later overwritten with h
static constexpr size_t OFF_AGGA   = OFF_BETA + SZ_BTH;
static constexpr size_t SZ_AGG     = (size_t)Bc * NC * Hc * 4;
static constexpr size_t OFF_AGGB   = OFF_AGGA + SZ_AGG;
static constexpr size_t OFF_START  = OFF_AGGB + SZ_AGG;
static constexpr size_t OFF_HNB    = OFF_START + SZ_AGG;
// total ~412 MB

typedef __attribute__((ext_vector_type(16))) __bf16 v16bf;
typedef __attribute__((ext_vector_type(8)))  float  v8f;

__device__ __forceinline__ unsigned short f2bf(float f) {
  unsigned int u = __float_as_uint(f);
  u += 0x7FFFu + ((u >> 16) & 1u);          // round-to-nearest-even
  return (unsigned short)(u >> 16);
}

// Load one 16-bit A/B fragment (16 rows x 32 K, row-major).
// lane&15 selects the row; kb = 8*(lane>=16). Two contiguous 8xbf16 runs.
__device__ __forceinline__ v16bf frag_ld(const unsigned short* rowp, int kb) {
  union { uint4 q[2]; v16bf v; } u;
  u.q[0] = *(const uint4*)(rowp + kb);
  u.q[1] = *(const uint4*)(rowp + kb + 16);
  return u.v;
}

// ---------------------------------------------------------------- fp32 -> bf16
__global__ void cvt_bf16_kernel(const float* __restrict__ src,
                                unsigned short* __restrict__ dst, int n) {
  int i = blockIdx.x * blockDim.x + threadIdx.x;
  int stride = gridDim.x * blockDim.x;
  for (; i < n; i += stride) dst[i] = f2bf(src[i]);
}

// ---------------------------------------------------------------- fused dual GEMM
// Block: 8 waves, one N-tile (16 cols of H), B strips (W_ih, W_gate) staged in LDS.
// Each wave: 4 M-tiles (64 rows). Grid covers (Mc/512) x (Hc/16) blocks.
__global__ void __launch_bounds__(256)
gemm_gates_kernel(const unsigned short* __restrict__ xb,
                  const unsigned short* __restrict__ wih,
                  const unsigned short* __restrict__ wg,
                  const float* __restrict__ b_ih,
                  const float* __restrict__ b_gate,
                  const float* __restrict__ log_a,
                  float* __restrict__ alpha,
                  float* __restrict__ beta) {
  __shared__ unsigned short sB[2 * 16 * LROW];   // ~66 KB (both weight strips)

  const int nTiles = Hc / 16;                    // 64
  const int n0    = (blockIdx.x % nTiles) * 16;
  const int mbase = (blockIdx.x / nTiles) * 512;
  const int waveId = threadIdx.x >> 5;
  const int lane   = threadIdx.x & 31;
  const int lr = lane & 15;
  const int kb = (lane >> 4) << 3;
  const int m0 = mbase + waveId * 64;

  // cooperative stage of W_ih / W_gate rows [n0, n0+16) x K=1024 into LDS
  {
    const unsigned short* g0 = wih + (size_t)n0 * Dc;
    const unsigned short* g1 = wg  + (size_t)n0 * Dc;
    for (int i = threadIdx.x; i < 16 * (Dc / 8); i += 256) {  // uint4 chunks
      int row = i >> 7, c8 = (i & 127) * 8;
      *(uint4*)(&sB[row * LROW + c8]) =
          *(const uint4*)(g0 + (size_t)row * Dc + c8);
      *(uint4*)(&sB[(16 + row) * LROW + c8]) =
          *(const uint4*)(g1 + (size_t)row * Dc + c8);
    }
  }
  __syncthreads();

  const unsigned short* a0 = xb + (size_t)(m0 +  0 + lr) * Dc;
  const unsigned short* a1 = xb + (size_t)(m0 + 16 + lr) * Dc;
  const unsigned short* a2 = xb + (size_t)(m0 + 32 + lr) * Dc;
  const unsigned short* a3 = xb + (size_t)(m0 + 48 + lr) * Dc;
  const unsigned short* sI = &sB[lr * LROW];
  const unsigned short* sG = &sB[(16 + lr) * LROW];

  v8f accI[4] = {}, accG[4] = {};
#pragma unroll 2
  for (int k0 = 0; k0 < Dc; k0 += 32) {
    v16bf bI = frag_ld(sI + k0, kb);
    v16bf bG = frag_ld(sG + k0, kb);
    v16bf fa0 = frag_ld(a0 + k0, kb);
    v16bf fa1 = frag_ld(a1 + k0, kb);
    v16bf fa2 = frag_ld(a2 + k0, kb);
    v16bf fa3 = frag_ld(a3 + k0, kb);
    accI[0] = __builtin_amdgcn_wmma_f32_16x16x32_bf16(false, fa0, false, bI, (short)0, accI[0], false, false);
    accG[0] = __builtin_amdgcn_wmma_f32_16x16x32_bf16(false, fa0, false, bG, (short)0, accG[0], false, false);
    accI[1] = __builtin_amdgcn_wmma_f32_16x16x32_bf16(false, fa1, false, bI, (short)0, accI[1], false, false);
    accG[1] = __builtin_amdgcn_wmma_f32_16x16x32_bf16(false, fa1, false, bG, (short)0, accG[1], false, false);
    accI[2] = __builtin_amdgcn_wmma_f32_16x16x32_bf16(false, fa2, false, bI, (short)0, accI[2], false, false);
    accG[2] = __builtin_amdgcn_wmma_f32_16x16x32_bf16(false, fa2, false, bG, (short)0, accG[2], false, false);
    accI[3] = __builtin_amdgcn_wmma_f32_16x16x32_bf16(false, fa3, false, bI, (short)0, accI[3], false, false);
    accG[3] = __builtin_amdgcn_wmma_f32_16x16x32_bf16(false, fa3, false, bG, (short)0, accG[3], false, false);
  }

  const int hi = lane >> 4;
  const int h  = n0 + lr;
  const float bih = b_ih[h];
  const float bg  = b_gate[h];
  const float ae  = __expf(log_a[h]);
#pragma unroll
  for (int t = 0; t < 4; ++t) {
#pragma unroll
    for (int r = 0; r < 8; ++r) {
      int m = m0 + t * 16 + hi * 8 + r;
      float ihv = accI[t][r] + bih;
      float gv  = accG[t][r] + bg;
      float g   = 1.f / (1.f + __expf(-gv));
      size_t idx = (size_t)m * Hc + h;
      alpha[idx] = (1.f - g) * ae;
      beta[idx]  = g * ihv;
    }
  }
}

// ---------------------------------------------------------------- scan pass 1: chunk aggregates
__global__ void scan_agg_kernel(const float* __restrict__ alpha,
                                const float* __restrict__ beta,
                                float* __restrict__ aggA,
                                float* __restrict__ aggB) {
  int g = blockIdx.x * blockDim.x + threadIdx.x;   // (b*NC + c)*H + h
  int h  = g & (Hc - 1);
  int bc = g >> 10;
  int c  = bc % NC;
  int b  = bc / NC;
  size_t idx = ((size_t)b * Tc + (size_t)c * CL) * Hc + h;
  float A = 1.f, Bv = 0.f;
  for (int t = 0; t < CL; ++t) {
    __builtin_prefetch(&alpha[idx + 16 * Hc], 0, 0);
    float a = alpha[idx], bv = beta[idx];
    Bv = a * Bv + bv;
    A *= a;
    idx += Hc;
  }
  aggA[g] = A;
  aggB[g] = Bv;
}

// ---------------------------------------------------------------- scan pass 2: prefix over chunks
__global__ void scan_prefix_kernel(const float* __restrict__ aggA,
                                   const float* __restrict__ aggB,
                                   float* __restrict__ startH) {
  int g = blockIdx.x * blockDim.x + threadIdx.x;   // b*H + h
  int h = g & (Hc - 1);
  int b = g >> 10;
  float hrun = 0.f;
  for (int c = 0; c < NC; ++c) {
    size_t idx = ((size_t)b * NC + c) * Hc + h;
    startH[idx] = hrun;
    hrun = aggA[idx] * hrun + aggB[idx];
  }
}

// ---------------------------------------------------------------- scan pass 3: apply (beta <- h)
__global__ void scan_apply_kernel(const float* __restrict__ alpha,
                                  float* __restrict__ beta,
                                  const float* __restrict__ startH) {
  int g = blockIdx.x * blockDim.x + threadIdx.x;
  int h  = g & (Hc - 1);
  int bc = g >> 10;
  int c  = bc % NC;
  int b  = bc / NC;
  size_t idx = ((size_t)b * Tc + (size_t)c * CL) * Hc + h;
  float hprev = startH[g];
  for (int t = 0; t < CL; ++t) {
    __builtin_prefetch(&alpha[idx + 16 * Hc], 0, 0);
    hprev = alpha[idx] * hprev + beta[idx];
    beta[idx] = hprev;
    idx += Hc;
  }
}

// ---------------------------------------------------------------- LayerNorm -> bf16
__global__ void __launch_bounds__(256)
ln_kernel(const float* __restrict__ hbuf,
          const float* __restrict__ ln_g,
          const float* __restrict__ ln_b,
          unsigned short* __restrict__ hnb) {
  __shared__ float ss[256], sq[256];
  const int row = blockIdx.x;
  const int tid = threadIdx.x;
  const float* hr = hbuf + (size_t)row * Hc;
  float v[4];
  float s = 0.f, q = 0.f;
#pragma unroll
  for (int i = 0; i < 4; ++i) {
    v[i] = hr[tid + i * 256];
    s += v[i];
    q += v[i] * v[i];
  }
  ss[tid] = s; sq[tid] = q;
  __syncthreads();
  for (int off = 128; off > 0; off >>= 1) {
    if (tid < off) { ss[tid] += ss[tid + off]; sq[tid] += sq[tid + off]; }
    __syncthreads();
  }
  const float mu  = ss[0] * (1.f / Hc);
  const float var = sq[0] * (1.f / Hc) - mu * mu;
  const float rstd = rsqrtf(var + LN_EPS);
  unsigned short* out = hnb + (size_t)row * Hc;
#pragma unroll
  for (int i = 0; i < 4; ++i) {
    int col = tid + i * 256;
    float hn = (v[i] - mu) * rstd * ln_g[col] + ln_b[col];
    out[col] = f2bf(hn);
  }
}

// ---------------------------------------------------------------- output GEMM (LDS-staged B strip)
__global__ void __launch_bounds__(256)
gemm_out_kernel(const unsigned short* __restrict__ hnb,
                const unsigned short* __restrict__ wo,
                const float* __restrict__ b_out,
                float* __restrict__ out) {
  __shared__ unsigned short sB[16 * LROW];       // ~33 KB

  const int nTiles = Oc / 16;
  const int n0    = (blockIdx.x % nTiles) * 16;
  const int mbase = (blockIdx.x / nTiles) * 512;
  const int waveId = threadIdx.x >> 5;
  const int lane   = threadIdx.x & 31;
  const int lr = lane & 15;
  const int kb = (lane >> 4) << 3;
  const int m0 = mbase + waveId * 64;

  {
    const unsigned short* g0 = wo + (size_t)n0 * Hc;
    for (int i = threadIdx.x; i < 16 * (Hc / 8); i += 256) {
      int row = i >> 7, c8 = (i & 127) * 8;
      *(uint4*)(&sB[row * LROW + c8]) =
          *(const uint4*)(g0 + (size_t)row * Hc + c8);
    }
  }
  __syncthreads();

  const unsigned short* a0 = hnb + (size_t)(m0 +  0 + lr) * Hc;
  const unsigned short* a1 = hnb + (size_t)(m0 + 16 + lr) * Hc;
  const unsigned short* a2 = hnb + (size_t)(m0 + 32 + lr) * Hc;
  const unsigned short* a3 = hnb + (size_t)(m0 + 48 + lr) * Hc;
  const unsigned short* sr = &sB[lr * LROW];

  v8f acc[4] = {};
#pragma unroll 2
  for (int k0 = 0; k0 < Hc; k0 += 32) {
    v16bf b  = frag_ld(sr + k0, kb);
    v16bf f0 = frag_ld(a0 + k0, kb);
    v16bf f1 = frag_ld(a1 + k0, kb);
    v16bf f2 = frag_ld(a2 + k0, kb);
    v16bf f3 = frag_ld(a3 + k0, kb);
    acc[0] = __builtin_amdgcn_wmma_f32_16x16x32_bf16(false, f0, false, b, (short)0, acc[0], false, false);
    acc[1] = __builtin_amdgcn_wmma_f32_16x16x32_bf16(false, f1, false, b, (short)0, acc[1], false, false);
    acc[2] = __builtin_amdgcn_wmma_f32_16x16x32_bf16(false, f2, false, b, (short)0, acc[2], false, false);
    acc[3] = __builtin_amdgcn_wmma_f32_16x16x32_bf16(false, f3, false, b, (short)0, acc[3], false, false);
  }

  const int hi = lane >> 4;
  const int o  = n0 + lr;
  const float bo = b_out[o];
#pragma unroll
  for (int t = 0; t < 4; ++t) {
#pragma unroll
    for (int r = 0; r < 8; ++r) {
      int m = m0 + t * 16 + hi * 8 + r;
      out[(size_t)m * Oc + o] = acc[t][r] + bo;
    }
  }
}

// ---------------------------------------------------------------- launcher
extern "C" void kernel_launch(void* const* d_in, const int* in_sizes, int n_in,
                              void* d_out, int out_size, void* d_ws, size_t ws_size,
                              hipStream_t stream) {
  (void)in_sizes; (void)n_in; (void)out_size; (void)ws_size;
  const float* x      = (const float*)d_in[0];
  const float* W_ih   = (const float*)d_in[1];
  const float* b_ih   = (const float*)d_in[2];
  const float* log_a  = (const float*)d_in[3];
  const float* W_gate = (const float*)d_in[4];
  const float* b_gate = (const float*)d_in[5];
  const float* W_out  = (const float*)d_in[6];
  const float* b_out  = (const float*)d_in[7];
  const float* ln_g   = (const float*)d_in[8];
  const float* ln_b   = (const float*)d_in[9];
  float* out = (float*)d_out;

  char* ws = (char*)d_ws;
  unsigned short* xb   = (unsigned short*)(ws + OFF_XB);
  unsigned short* wihb = (unsigned short*)(ws + OFF_WIHB);
  unsigned short* wgb  = (unsigned short*)(ws + OFF_WGB);
  unsigned short* wob  = (unsigned short*)(ws + OFF_WOB);
  float* alpha  = (float*)(ws + OFF_ALPHA);
  float* beta   = (float*)(ws + OFF_BETA);
  float* aggA   = (float*)(ws + OFF_AGGA);
  float* aggB   = (float*)(ws + OFF_AGGB);
  float* startH = (float*)(ws + OFF_START);
  unsigned short* hnb = (unsigned short*)(ws + OFF_HNB);

  // 1) bf16 staging
  cvt_bf16_kernel<<<4096, 256, 0, stream>>>(x, xb, Mc * Dc);
  cvt_bf16_kernel<<<1024, 256, 0, stream>>>(W_ih, wihb, Hc * Dc);
  cvt_bf16_kernel<<<1024, 256, 0, stream>>>(W_gate, wgb, Hc * Dc);
  cvt_bf16_kernel<<<1024, 256, 0, stream>>>(W_out, wob, Oc * Hc);

  // 2) fused dual GEMM -> alpha, beta  (block = 512 rows x 16 cols)
  gemm_gates_kernel<<<(Mc / 512) * (Hc / 16), 256, 0, stream>>>(
      xb, wihb, wgb, b_ih, b_gate, log_a, alpha, beta);

  // 3) chunked scan
  scan_agg_kernel<<<(Bc * NC * Hc) / 256, 256, 0, stream>>>(alpha, beta, aggA, aggB);
  scan_prefix_kernel<<<(Bc * Hc) / 256, 256, 0, stream>>>(aggA, aggB, startH);
  scan_apply_kernel<<<(Bc * NC * Hc) / 256, 256, 0, stream>>>(alpha, beta, startH);

  // 4) LayerNorm -> bf16 hn
  ln_kernel<<<Mc, 256, 0, stream>>>(beta, ln_g, ln_b, hnb);

  // 5) output GEMM
  gemm_out_kernel<<<(Mc / 512) * (Oc / 16), 256, 0, stream>>>(hnb, wob, b_out, out);
}
